// HistoricCurrent_69114613727328
// MI455X (gfx1250) — matrix operand order repaired and checked by reading, not compiled, not run on hardware
//
#include <hip/hip_runtime.h>
#include <hip/hip_bf16.h>
#include <math.h>

#define B_  2048
#define S_  100
#define E_  768
#define H_  512

typedef __attribute__((ext_vector_type(16))) __bf16 bf16x16;
typedef __attribute__((ext_vector_type(8)))  __bf16 bf16x8;
typedef __attribute__((ext_vector_type(8)))  float  f32x8;

union Frag16 { bf16x16 v; bf16x8 h[2]; };

// A-fragment (16x32 bf16, row-major source): lane L -> row M = L%16, half = L/16.
// elems 0..7  = K[kb + half*8 .. +7]        (one b128)
// elems 8..15 = K[kb + 16 + half*8 .. +7]   (one b128)
__device__ __forceinline__ bf16x16 load_a_frag(const __bf16* rowptr, int kb, int half) {
  Frag16 f;
  f.h[0] = *(const bf16x8*)(rowptr + kb + half * 8);
  f.h[1] = *(const bf16x8*)(rowptr + kb + 16 + half * 8);
  return f.v;
}

// B-fragment from pre-packed buffer: 512 bf16 per (ktile,ntile) frag, lane-major,
// so each lane reads 16 contiguous bf16 (32 bytes).
__device__ __forceinline__ bf16x16 load_b_frag(const __bf16* pack, int frag, int lane) {
  return *(const bf16x16*)(pack + (size_t)frag * 512 + lane * 16);
}

__device__ __forceinline__ f32x8 wmma_bf16(bf16x16 a, bf16x16 b, f32x8 c) {
  return __builtin_amdgcn_wmma_f32_16x16x32_bf16(false, a, false, b, (short)0, c, false, false);
}

__device__ __forceinline__ float sigm(float x) { return 1.0f / (1.0f + expf(-x)); }

// ---------------- utility kernels ----------------

__global__ void cvt_bf16_kernel(const float* __restrict__ in, __bf16* __restrict__ out, size_t n) {
  size_t i = (size_t)blockIdx.x * blockDim.x + threadIdx.x;
  size_t stride = (size_t)gridDim.x * blockDim.x;
  for (; i < n; i += stride) out[i] = (__bf16)in[i];
}

__global__ void zero_u32_kernel(unsigned* __restrict__ p, size_t n) {
  size_t i = (size_t)blockIdx.x * blockDim.x + threadIdx.x;
  size_t stride = (size_t)gridDim.x * blockDim.x;
  for (; i < n; i += stride) p[i] = 0u;
}

// Pack row-major fp32 W[K][N] into bf16 WMMA-B fragment order.
__global__ void pack_w_kernel(const float* __restrict__ W, __bf16* __restrict__ out, int K, int N) {
  int idx = blockIdx.x * blockDim.x + threadIdx.x;
  if (idx >= K * N) return;
  int k = idx / N, n = idx % N;
  int kt = k >> 5, kk = k & 31, ntile = n >> 4, nn = n & 15;
  int hf = (kk >> 3) & 1, e = (kk & 7) + ((kk >> 4) << 3);
  int lane = hf * 16 + nn;
  int frag = kt * (N >> 4) + ntile;
  out[(size_t)frag * 512 + lane * 16 + e] = (__bf16)W[idx];
}

__global__ void bias_sum_kernel(const float* __restrict__ a, const float* __restrict__ b,
                                float* __restrict__ o, int n) {
  int i = blockIdx.x * blockDim.x + threadIdx.x;
  if (i < n) o[i] = a[i] + b[i];
}

// hist mean -> combined[:, 512:1024]
__global__ void mean_kernel(const float* __restrict__ hsum, __bf16* __restrict__ combined) {
  int idx = blockIdx.x * blockDim.x + threadIdx.x;  // B_*H_
  if (idx >= B_ * H_) return;
  int b = idx >> 9, j = idx & 511;
  combined[(size_t)b * 1024 + 512 + j] = (__bf16)(hsum[idx] * (1.0f / (float)S_));
}

// ---------------- fused Time-LSTM step ----------------
// Per step s:
//   cs1 = tanh(c @ W_d + b_d); c_adj = c - cs1 + cs1*t
//   gates = h @ W_all + x_s @ U_all + (W_all_b + U_all_b)
//   c' = sig(f)*c_adj + sig(i)*sig(ct);  h' = sig(o)*tanh(c')
// Tiling: 8 waves/WG; wave -> 32 rows x 16 cols (2 C-tiles share B-frags).
// WG tile: 64 rows x 64 cols of H. Grid: (2048/64, 512/64) = (32, 8).
__global__ __launch_bounds__(256)
void step_kernel(const float*  __restrict__ c_f32_in,
                 const __bf16* __restrict__ c_bf_in,
                 const __bf16* __restrict__ h_bf_in,
                 const __bf16* __restrict__ xhist,
                 const float*  __restrict__ tstamp,
                 const __bf16* __restrict__ WdP,
                 const __bf16* __restrict__ WallP,
                 const __bf16* __restrict__ UP,
                 const float*  __restrict__ bd,
                 const float*  __restrict__ ball,
                 float*  __restrict__ c_f32_out,
                 __bf16* __restrict__ c_bf_out,
                 __bf16* __restrict__ h_bf_out,
                 float*  __restrict__ hsum,
                 int s)
{
  const int tid  = threadIdx.x;
  const int lane = tid & 31, w = tid >> 5;
  const int wr = w >> 2, wc = w & 3;
  const int half = lane >> 4, lm = lane & 15;
  const int rowWG = blockIdx.x * 64;
  const int cb    = blockIdx.y * 64 + wc * 16;   // column base within H
  const int nt    = cb >> 4;                     // n-tile index within H (0..31)

  f32x8 accd[2];
  f32x8 accg[4][2];
#pragma unroll
  for (int su = 0; su < 2; ++su) {
    accd[su] = (f32x8)0.0f;
#pragma unroll
    for (int g = 0; g < 4; ++g) accg[g][su] = (f32x8)0.0f;
  }

  const int r0 = rowWG + wr * 32 + lm;   // this lane's A row, sub-tile 0
  const int r1 = r0 + 16;                // sub-tile 1
  const __bf16* cA0 = c_bf_in + (size_t)r0 * H_;
  const __bf16* cA1 = c_bf_in + (size_t)r1 * H_;
  const __bf16* hA0 = h_bf_in + (size_t)r0 * H_;
  const __bf16* hA1 = h_bf_in + (size_t)r1 * H_;

  // ---- K loop over H: decay GEMM (c @ W_d) + recurrent gates (h @ W_all) ----
  for (int kt = 0; kt < H_ / 32; ++kt) {
    const int kb = kt * 32;
    bf16x16 ac0 = load_a_frag(cA0, kb, half);
    bf16x16 ac1 = load_a_frag(cA1, kb, half);
    bf16x16 ah0 = load_a_frag(hA0, kb, half);
    bf16x16 ah1 = load_a_frag(hA1, kb, half);
    bf16x16 bdF = load_b_frag(WdP, kt * 32 + nt, lane);
    accd[0] = wmma_bf16(ac0, bdF, accd[0]);
    accd[1] = wmma_bf16(ac1, bdF, accd[1]);
#pragma unroll
    for (int g = 0; g < 4; ++g) {
      bf16x16 bg = load_b_frag(WallP, kt * 128 + g * 32 + nt, lane);
      accg[g][0] = wmma_bf16(ah0, bg, accg[g][0]);
      accg[g][1] = wmma_bf16(ah1, bg, accg[g][1]);
    }
  }

  // ---- K loop over E: fused input projection (x_s @ U_all) ----
  const __bf16* xA0 = xhist + ((size_t)r0 * S_ + s) * E_;
  const __bf16* xA1 = xhist + ((size_t)r1 * S_ + s) * E_;
  for (int kt = 0; kt < E_ / 32; ++kt) {
    const int kb = kt * 32;
    bf16x16 ax0 = load_a_frag(xA0, kb, half);
    bf16x16 ax1 = load_a_frag(xA1, kb, half);
#pragma unroll
    for (int g = 0; g < 4; ++g) {
      bf16x16 bg = load_b_frag(UP, kt * 128 + g * 32 + nt, lane);
      accg[g][0] = wmma_bf16(ax0, bg, accg[g][0]);
      accg[g][1] = wmma_bf16(ax1, bg, accg[g][1]);
    }
  }

  // ---- elementwise gate epilogue (fp32) ----
  const int j = cb + lm;
  const float bdj = bd[j];
  const float bf_ = ball[j], bi_ = ball[H_ + j], bo_ = ball[2 * H_ + j], bc_ = ball[3 * H_ + j];
#pragma unroll
  for (int su = 0; su < 2; ++su) {
    const int rbase = rowWG + wr * 32 + su * 16 + half * 8;  // C-layout: VGPR i -> row rbase+i
#pragma unroll
    for (int i = 0; i < 8; ++i) {
      const int row = rbase + i;
      const size_t idx = (size_t)row * H_ + j;
      const float cold = c_f32_in[idx];
      const float t    = tstamp[row * S_ + s];
      const float cs1  = tanhf(accd[su][i] + bdj);
      const float cadj = cold - cs1 + cs1 * t;
      const float f  = sigm(accg[0][su][i] + bf_);
      const float ig = sigm(accg[1][su][i] + bi_);
      const float o  = sigm(accg[2][su][i] + bo_);
      const float ct = sigm(accg[3][su][i] + bc_);
      const float cn = f * cadj + ig * ct;
      const float hn = o * tanhf(cn);
      c_f32_out[idx] = cn;
      c_bf_out[idx]  = (__bf16)cn;
      h_bf_out[idx]  = (__bf16)hn;
      hsum[idx]     += hn;
    }
  }
}

// ---------------- generic M=2048, N=512 WMMA GEMM + bias + ReLU -> bf16 ----------------
__global__ __launch_bounds__(256)
void gemm_relu_kernel(const __bf16* __restrict__ A, int lda,
                      const __bf16* __restrict__ BP, const float* __restrict__ bias,
                      int K, __bf16* __restrict__ out, int ldo)
{
  const int tid  = threadIdx.x;
  const int lane = tid & 31, w = tid >> 5;
  const int wr = w >> 2, wc = w & 3;
  const int half = lane >> 4, lm = lane & 15;
  const int rowWG = blockIdx.x * 64;
  const int cb    = blockIdx.y * 64 + wc * 16;
  const int nt    = cb >> 4;

  f32x8 acc[2];
  acc[0] = (f32x8)0.0f; acc[1] = (f32x8)0.0f;

  const int r0 = rowWG + wr * 32 + lm;
  const __bf16* A0 = A + (size_t)r0 * lda;
  const __bf16* A1 = A + (size_t)(r0 + 16) * lda;

  for (int kt = 0; kt < (K >> 5); ++kt) {
    const int kb = kt * 32;
    bf16x16 a0 = load_a_frag(A0, kb, half);
    bf16x16 a1 = load_a_frag(A1, kb, half);
    bf16x16 b  = load_b_frag(BP, kt * 32 + nt, lane);  // N=512 -> 32 ntiles
    acc[0] = wmma_bf16(a0, b, acc[0]);
    acc[1] = wmma_bf16(a1, b, acc[1]);
  }

  const int j = cb + lm;
  const float bj = bias[j];
#pragma unroll
  for (int su = 0; su < 2; ++su) {
    const int rbase = rowWG + wr * 32 + su * 16 + half * 8;
#pragma unroll
    for (int i = 0; i < 8; ++i) {
      float v = acc[su][i] + bj;
      v = v > 0.0f ? v : 0.0f;
      out[(size_t)(rbase + i) * ldo + j] = (__bf16)v;
    }
  }
}

// ---------------- final N=2 layer (plain dot product) ----------------
__global__ void final_kernel(const __bf16* __restrict__ x, const float* __restrict__ fw,
                             const float* __restrict__ fb, float* __restrict__ out)
{
  int idx = blockIdx.x * blockDim.x + threadIdx.x;  // B_*2
  if (idx >= B_ * 2) return;
  int b = idx >> 1, c = idx & 1;
  float acc = fb[c];
  const __bf16* xr = x + (size_t)b * H_;
  for (int k = 0; k < H_; ++k) acc += (float)xr[k] * fw[k * 2 + c];
  out[idx] = acc;
}

// ---------------- host ----------------

extern "C" void kernel_launch(void* const* d_in, const int* in_sizes, int n_in,
                              void* d_out, int out_size, void* d_ws, size_t ws_size,
                              hipStream_t stream) {
  const float* tweet   = (const float*)d_in[0];
  const float* hist    = (const float*)d_in[1];
  /* d_in[2] = lens: unused by reference */
  const float* tstamp  = (const float*)d_in[3];
  const float* Wall_w  = (const float*)d_in[4];
  const float* Wall_b  = (const float*)d_in[5];
  const float* U_w     = (const float*)d_in[6];
  const float* U_b     = (const float*)d_in[7];
  const float* Wd_w    = (const float*)d_in[8];
  const float* Wd_b    = (const float*)d_in[9];
  const float* fcct_w  = (const float*)d_in[10];
  const float* fcct_b  = (const float*)d_in[11];
  const float* fccat_w = (const float*)d_in[12];
  const float* fccat_b = (const float*)d_in[13];
  const float* fin_w   = (const float*)d_in[14];
  const float* fin_b   = (const float*)d_in[15];

  char* ws = (char*)d_ws;
  size_t off = 0;
  auto alloc = [&](size_t bytes) -> void* {
    void* p = ws + off;
    off += (bytes + 255) & ~(size_t)255;
    return p;
  };

  __bf16* hist_bf   = (__bf16*)alloc((size_t)B_ * S_ * E_ * 2);   // 314.6 MB
  __bf16* tweet_bf  = (__bf16*)alloc((size_t)B_ * E_ * 2);
  __bf16* UP        = (__bf16*)alloc((size_t)E_ * 4 * H_ * 2);
  __bf16* WallP     = (__bf16*)alloc((size_t)H_ * 4 * H_ * 2);
  __bf16* WdP       = (__bf16*)alloc((size_t)H_ * H_ * 2);
  __bf16* fcctP     = (__bf16*)alloc((size_t)E_ * H_ * 2);
  __bf16* fccatP    = (__bf16*)alloc((size_t)2 * H_ * H_ * 2);
  float*  ball      = (float*)alloc((size_t)4 * H_ * 4);
  float*  c_f32[2]  = { (float*)alloc((size_t)B_ * H_ * 4),  (float*)alloc((size_t)B_ * H_ * 4) };
  __bf16* c_bf[2]   = { (__bf16*)alloc((size_t)B_ * H_ * 2), (__bf16*)alloc((size_t)B_ * H_ * 2) };
  __bf16* h_bf[2]   = { (__bf16*)alloc((size_t)B_ * H_ * 2), (__bf16*)alloc((size_t)B_ * H_ * 2) };
  float*  hsum      = (float*)alloc((size_t)B_ * H_ * 4);
  __bf16* combined  = (__bf16*)alloc((size_t)B_ * 2 * H_ * 2);
  __bf16* xact      = (__bf16*)alloc((size_t)B_ * H_ * 2);

  const int TB = 256;

  // Prologue: conversions, weight packing, bias sum, state zero-init.
  cvt_bf16_kernel<<<16384, TB, 0, stream>>>(hist,  hist_bf,  (size_t)B_ * S_ * E_);
  cvt_bf16_kernel<<<2048,  TB, 0, stream>>>(tweet, tweet_bf, (size_t)B_ * E_);

  pack_w_kernel<<<(E_ * 4 * H_ + TB - 1) / TB, TB, 0, stream>>>(U_w,     UP,     E_,    4 * H_);
  pack_w_kernel<<<(H_ * 4 * H_ + TB - 1) / TB, TB, 0, stream>>>(Wall_w,  WallP,  H_,    4 * H_);
  pack_w_kernel<<<(H_ * H_     + TB - 1) / TB, TB, 0, stream>>>(Wd_w,    WdP,    H_,    H_);
  pack_w_kernel<<<(E_ * H_     + TB - 1) / TB, TB, 0, stream>>>(fcct_w,  fcctP,  E_,    H_);
  pack_w_kernel<<<(2 * H_ * H_ + TB - 1) / TB, TB, 0, stream>>>(fccat_w, fccatP, 2*H_,  H_);

  bias_sum_kernel<<<(4 * H_ + TB - 1) / TB, TB, 0, stream>>>(Wall_b, U_b, ball, 4 * H_);

  zero_u32_kernel<<<2048, TB, 0, stream>>>((unsigned*)hsum,     (size_t)B_ * H_);
  zero_u32_kernel<<<2048, TB, 0, stream>>>((unsigned*)c_f32[0], (size_t)B_ * H_);
  zero_u32_kernel<<<1024, TB, 0, stream>>>((unsigned*)c_bf[0],  (size_t)B_ * H_ / 2);
  zero_u32_kernel<<<1024, TB, 0, stream>>>((unsigned*)h_bf[0],  (size_t)B_ * H_ / 2);

  // tweet branch: combined[:, 0:512] = relu(tweet @ fc_ct + b)
  {
    dim3 grid(B_ / 64, H_ / 64);
    gemm_relu_kernel<<<grid, TB, 0, stream>>>(tweet_bf, E_, fcctP, fcct_b, E_, combined, 2 * H_);
  }

  // Time-LSTM scan: 100 fused step kernels, ping-pong state.
  {
    dim3 grid(B_ / 64, H_ / 64);
    for (int s = 0; s < S_; ++s) {
      int in = s & 1, outb = (s + 1) & 1;
      step_kernel<<<grid, TB, 0, stream>>>(c_f32[in], c_bf[in], h_bf[in],
                                           hist_bf, tstamp,
                                           WdP, WallP, UP, Wd_b, ball,
                                           c_f32[outb], c_bf[outb], h_bf[outb],
                                           hsum, s);
    }
  }

  // combined[:, 512:1024] = mean_s(h_s)
  mean_kernel<<<(B_ * H_ + TB - 1) / TB, TB, 0, stream>>>(hsum, combined);

  // x = relu(combined @ fc_concat + b)
  {
    dim3 grid(B_ / 64, H_ / 64);
    gemm_relu_kernel<<<grid, TB, 0, stream>>>(combined, 2 * H_, fccatP, fccat_b, 2 * H_, xact, H_);
  }

  // out = x @ final_w + final_b
  final_kernel<<<(B_ * 2 + TB - 1) / TB, TB, 0, stream>>>(xact, fin_w, fin_b, (float*)d_out);
}
